// MambaBlock_25598005084283
// MI455X (gfx1250) — matrix operand order repaired and checked by reading, not compiled, not run on hardware
//
#include <hip/hip_runtime.h>
#include <hip/hip_bf16.h>
#include <math.h>

// ---------------------------------------------------------------------------
// MambaBlock on gfx1250 (MI455X): bf16 WMMA GEMMs + wave32 parallel SSM scan
//   k_inproj_silu : xs = (x@W_in[0:E].T) * sigmoid(x@W_in[E:2E].T)   [WMMA]
//   k_conv_silu   : u  = silu(causal depthwise conv1d(xs, K=4))
//   k_delta       : delta = sigmoid(u @ Bp)                           [WMMA]
//   k_scan        : h = exp(-exp(A)*d)*h + d*Bp*u ; y = <h,Cp> + Dp*u
//   k_outproj     : out = y @ W_out.T + x                             [WMMA]
// Sigmoids use v_rcp_f32 (1 ULP) instead of IEEE division.
// ---------------------------------------------------------------------------

typedef __attribute__((ext_vector_type(16))) __bf16 v16bf;
typedef __attribute__((ext_vector_type(8)))  float  v8f;

#define DIMX   512
#define DSTATE 16
#define EDIM   1024
#define BB     4
#define LL     2048
#define MTOT   (BB * LL)     // 8192 rows (b,l flattened)
#define LDSS   40            // LDS row stride in ushorts (32 + 8 pad)

__device__ __forceinline__ unsigned short f32_bf16(float f) {
  unsigned int x = __float_as_uint(f);
  x += 0x7FFFu + ((x >> 16) & 1u);          // round-to-nearest-even
  return (unsigned short)(x >> 16);
}

__device__ __forceinline__ float fast_sigmoid(float g) {
  // v_exp_f32 + v_add + v_rcp_f32 (no IEEE div sequence)
  return __builtin_amdgcn_rcpf(1.0f + __expf(-g));
}

union BF16Frag { unsigned short u[16]; v16bf v; };

// A-matrix 16x32 bf16 fragment (ISA 7.12.2): lanes 0-15 hold M=0..15 with
// K=0..7 (v0..v3) and K=16..23 (v4..v7); lanes 16-31 hold K=8..15 / K=24..31.
__device__ __forceinline__ v16bf load_a_frag(const unsigned short* base, int lane) {
  BF16Frag f;
  const int m = lane & 15, hi = lane >> 4;
  const unsigned short* r = base + m * LDSS;
#pragma unroll
  for (int j = 0; j < 4; ++j) {
    f.u[2 * j]     = r[hi * 8 + 2 * j];
    f.u[2 * j + 1] = r[hi * 8 + 2 * j + 1];
    f.u[8 + 2 * j]     = r[16 + hi * 8 + 2 * j];
    f.u[8 + 2 * j + 1] = r[16 + hi * 8 + 2 * j + 1];
  }
  return f.v;
}

// B-matrix 32x16 bf16 fragment: lane = N column; lanes 0-15 hold K=0..7 and
// K=16..23, lanes 16-31 hold K=8..15 / K=24..31. LDS tile stored [N][K].
__device__ __forceinline__ v16bf load_b_frag(const unsigned short* base, int lane) {
  BF16Frag f;
  const int n = lane & 15, hi = lane >> 4;
  const unsigned short* r = base + n * LDSS;
#pragma unroll
  for (int j = 0; j < 8; ++j) {
    f.u[2 * j]     = r[hi * 8 + 2 * j + (j >= 4 ? 8 : 0)];
    f.u[2 * j + 1] = r[hi * 8 + 2 * j + 1 + (j >= 4 ? 8 : 0)];
  }
  return f.v;
}

// ---------------- in_proj + SiLU gate --------------------------------------
__global__ __launch_bounds__(256)
void k_inproj_silu(const float* __restrict__ x, const float* __restrict__ W_in,
                   float* __restrict__ xs) {
  __shared__ unsigned short As[64 * LDSS];
  __shared__ unsigned short Bs[2][64 * LDSS];
  const int tid = threadIdx.x, lane = tid & 31, wid = tid >> 5;
  const int wm = wid >> 1, wn = wid & 1;            // 4x2 wave grid
  const int m0 = blockIdx.x * 64, n0 = blockIdx.y * 64;
  v8f acc[2][2] = {};

  for (int kb = 0; kb < DIMX; kb += 32) {
    __syncthreads();
#pragma unroll
    for (int i = 0; i < 8; ++i) {                   // A tile 64x32 fp32->bf16
      int f = tid * 8 + i, r = f >> 5, c = f & 31;
      As[r * LDSS + c] = f32_bf16(x[(m0 + r) * DIMX + kb + c]);
    }
#pragma unroll
    for (int h = 0; h < 2; ++h) {                   // B tiles: xs half, gate half
#pragma unroll
      for (int i = 0; i < 8; ++i) {
        int f = tid * 8 + i, n = f >> 5, c = f & 31;
        Bs[h][n * LDSS + c] = f32_bf16(W_in[(h * EDIM + n0 + n) * DIMX + kb + c]);
      }
    }
    __syncthreads();

    v16bf a = load_a_frag(&As[(wm * 16) * LDSS], lane);
#pragma unroll
    for (int t = 0; t < 2; ++t) {
      v16bf b0 = load_b_frag(&Bs[0][(wn * 32 + t * 16) * LDSS], lane);
      acc[0][t] = __builtin_amdgcn_wmma_f32_16x16x32_bf16(
          false, a, false, b0, (short)0, acc[0][t], false, false);
      v16bf b1 = load_b_frag(&Bs[1][(wn * 32 + t * 16) * LDSS], lane);
      acc[1][t] = __builtin_amdgcn_wmma_f32_16x16x32_bf16(
          false, a, false, b1, (short)0, acc[1][t], false, false);
    }
  }

  const int n = lane & 15, hi = lane >> 4;
#pragma unroll
  for (int t = 0; t < 2; ++t)
#pragma unroll
    for (int r = 0; r < 8; ++r) {
      int gm = m0 + wm * 16 + hi * 8 + r;
      int gc = n0 + wn * 32 + t * 16 + n;
      float s = acc[0][t][r], g = acc[1][t][r];
      xs[gm * EDIM + gc] = s * fast_sigmoid(g);
    }
}

// ---------------- causal depthwise conv (K=4) + SiLU -----------------------
__global__ __launch_bounds__(256)
void k_conv_silu(const float* __restrict__ xs, const float* __restrict__ Wc,
                 float* __restrict__ u) {
  const int idx = blockIdx.x * 256 + threadIdx.x;   // (b,l,e) flat
  const int e = idx & (EDIM - 1);
  const int m = idx >> 10;                          // b*L + l
  const int l = m & (LL - 1);
  const float w0 = Wc[e * 4 + 0], w1 = Wc[e * 4 + 1];
  const float w2 = Wc[e * 4 + 2], w3 = Wc[e * 4 + 3];
  const float* col = xs + (size_t)(m - l) * EDIM + e;   // start of this batch
  float acc = col[(size_t)l * EDIM] * w3;
  if (l >= 1) acc += col[(size_t)(l - 1) * EDIM] * w2;
  if (l >= 2) acc += col[(size_t)(l - 2) * EDIM] * w1;
  if (l >= 3) acc += col[(size_t)(l - 3) * EDIM] * w0;
  u[idx] = acc * fast_sigmoid(acc);                 // silu
}

// ---------------- delta = sigmoid(u @ Bp)  (skinny WMMA, N=16) -------------
__global__ __launch_bounds__(256)
void k_delta(const float* __restrict__ u, const float* __restrict__ Bp,
             float* __restrict__ delta) {
  __shared__ unsigned short Us[128 * LDSS];
  __shared__ unsigned short Bps[16 * LDSS];
  const int tid = threadIdx.x, lane = tid & 31, wid = tid >> 5;
  const int m0 = blockIdx.x * 128;                  // 8 waves x 16 rows
  v8f acc = {};

  for (int kb = 0; kb < EDIM; kb += 32) {
    __syncthreads();
#pragma unroll
    for (int i = 0; i < 16; ++i) {                  // A tile 128x32
      int f = tid * 16 + i, r = f >> 5, c = f & 31;
      Us[r * LDSS + c] = f32_bf16(u[(m0 + r) * EDIM + kb + c]);
    }
#pragma unroll
    for (int i = 0; i < 2; ++i) {                   // Bp slice, transposed [n][k]
      int f = tid * 2 + i, n = f >> 5, k = f & 31;
      Bps[n * LDSS + k] = f32_bf16(Bp[(kb + k) * DSTATE + n]);
    }
    __syncthreads();
    v16bf a = load_a_frag(&Us[(wid * 16) * LDSS], lane);
    v16bf b = load_b_frag(&Bps[0], lane);
    acc = __builtin_amdgcn_wmma_f32_16x16x32_bf16(
        false, a, false, b, (short)0, acc, false, false);
  }

  const int n = lane & 15, hi = lane >> 4;
#pragma unroll
  for (int r = 0; r < 8; ++r) {
    int gm = m0 + wid * 16 + hi * 8 + r;
    delta[gm * DSTATE + n] = fast_sigmoid(acc[r]);
  }
}

// ---------------- selective-scan: lane per (b,e,n) state -------------------
__global__ __launch_bounds__(256)
void k_scan(const float* __restrict__ u, const float* __restrict__ delta,
            const float* __restrict__ A, const float* __restrict__ Bp,
            const float* __restrict__ Cp, const float* __restrict__ Dp,
            float* __restrict__ y) {
  __shared__ float sD[256];                          // 16 steps x 16 n
  __shared__ float sU[256];                          // 16 steps x 16 e
  const int tid = threadIdx.x;
  const int n  = tid & 15;                           // state index
  const int el = tid >> 4;                           // local channel 0..15
  const int b  = blockIdx.x >> 6;
  const int e0 = (blockIdx.x & 63) << 4;
  const int e  = e0 + el;
  const float aneg = -__expf(A[e * DSTATE + n]);
  const float bpn  = Bp[e * DSTATE + n];
  const float cpn  = Cp[e * DSTATE + n];
  const float dpe  = Dp[e];
  float h = 0.0f;
  const float* ub = u     + (size_t)b * LL * EDIM;
  const float* db = delta + (size_t)b * LL * DSTATE;
  float*       yb = y     + (size_t)b * LL * EDIM;

  for (int lc = 0; lc < LL; lc += 16) {
    __syncthreads();
    sD[tid] = db[lc * DSTATE + tid];                 // 16 steps of delta
    { int s = tid >> 4, j = tid & 15;
      sU[tid] = ub[(lc + s) * EDIM + e0 + j]; }      // 16 steps of u
    __syncthreads();
    if (lc + 16 < LL)                                // global_prefetch_b8
      __builtin_prefetch(&db[(lc + 16) * DSTATE + tid], 0, 0);
#pragma unroll
    for (int s = 0; s < 16; ++s) {
      float dl = sD[s * DSTATE + n];
      float du = sU[s * 16 + el];
      float at = __expf(aneg * dl);                  // A_bar (h-independent)
      h = at * h + dl * bpn * du;                    // the only serial FMA
      float yv = h * cpn;
      yv += __shfl_xor(yv, 1, 32);                   // reduce 16 n-lanes
      yv += __shfl_xor(yv, 2, 32);
      yv += __shfl_xor(yv, 4, 32);
      yv += __shfl_xor(yv, 8, 32);
      if (n == 0) yb[(lc + s) * EDIM + e] = yv + dpe * du;
    }
  }
}

// ---------------- out_proj + residual --------------------------------------
__global__ __launch_bounds__(256)
void k_outproj(const float* __restrict__ y, const float* __restrict__ W_out,
               const float* __restrict__ xres, float* __restrict__ out) {
  __shared__ unsigned short As[64 * LDSS];
  __shared__ unsigned short Bs[64 * LDSS];
  const int tid = threadIdx.x, lane = tid & 31, wid = tid >> 5;
  const int wm = wid >> 1, wn = wid & 1;
  const int m0 = blockIdx.x * 64, n0 = blockIdx.y * 64;
  v8f acc[2] = {};

  for (int kb = 0; kb < EDIM; kb += 32) {
    __syncthreads();
#pragma unroll
    for (int i = 0; i < 8; ++i) {
      int f = tid * 8 + i, r = f >> 5, c = f & 31;
      As[r * LDSS + c] = f32_bf16(y[(m0 + r) * EDIM + kb + c]);
      Bs[r * LDSS + c] = f32_bf16(W_out[(n0 + r) * EDIM + kb + c]);
    }
    __syncthreads();
    v16bf a = load_a_frag(&As[(wm * 16) * LDSS], lane);
#pragma unroll
    for (int t = 0; t < 2; ++t) {
      v16bf b = load_b_frag(&Bs[(wn * 32 + t * 16) * LDSS], lane);
      acc[t] = __builtin_amdgcn_wmma_f32_16x16x32_bf16(
          false, a, false, b, (short)0, acc[t], false, false);
    }
  }

  const int n = lane & 15, hi = lane >> 4;
#pragma unroll
  for (int t = 0; t < 2; ++t)
#pragma unroll
    for (int r = 0; r < 8; ++r) {
      int gm = m0 + wm * 16 + hi * 8 + r;
      int gc = n0 + wn * 32 + t * 16 + n;
      out[gm * DIMX + gc] = acc[t][r] + xres[gm * DIMX + gc];
    }
}

// ---------------------------------------------------------------------------
extern "C" void kernel_launch(void* const* d_in, const int* in_sizes, int n_in,
                              void* d_out, int out_size, void* d_ws, size_t ws_size,
                              hipStream_t stream) {
  (void)in_sizes; (void)n_in; (void)out_size; (void)ws_size;
  const float* x      = (const float*)d_in[0];
  const float* W_in   = (const float*)d_in[1];
  const float* W_conv = (const float*)d_in[2];
  const float* A      = (const float*)d_in[3];
  const float* Bp     = (const float*)d_in[4];
  const float* Cp     = (const float*)d_in[5];
  const float* Dp     = (const float*)d_in[6];
  const float* W_out  = (const float*)d_in[7];
  float* out = (float*)d_out;

  char* ws = (char*)d_ws;
  const size_t NBLE = (size_t)MTOT * EDIM * sizeof(float);   // 32 MB
  float* xs = (float*)(ws);                                  // [M,E]
  float* u  = (float*)(ws + NBLE);                           // [M,E]
  float* dl = (float*)(ws + 2 * NBLE);                       // [M,16]
  float* yy = (float*)(ws + 2 * NBLE + (size_t)MTOT * DSTATE * sizeof(float));

  k_inproj_silu<<<dim3(MTOT / 64, EDIM / 64), 256, 0, stream>>>(x, W_in, xs);
  k_conv_silu<<<(MTOT * EDIM) / 256, 256, 0, stream>>>(xs, W_conv, u);
  k_delta<<<MTOT / 128, 256, 0, stream>>>(u, Bp, dl);
  k_scan<<<BB * (EDIM / 16), 256, 0, stream>>>(u, dl, A, Bp, Cp, Dp, yy);
  k_outproj<<<dim3(MTOT / 64, DIMX / 64), 256, 0, stream>>>(yy, W_out, x, out);
}
